// geo_CGNN_15272903704942
// MI455X (gfx1250) — matrix operand order repaired and checked by reading, not compiled, not run on hardware
//
#include <hip/hip_runtime.h>

// ---------------------------------------------------------------------------
// geo-CGNN forward on MI455X (gfx1250), wave32 + WMMA f32<-f16 16x16x32.
// ---------------------------------------------------------------------------

typedef __attribute__((ext_vector_type(16))) _Float16 v16h;
typedef __attribute__((ext_vector_type(8)))  _Float16 v8h;
typedef __attribute__((ext_vector_type(8)))  float    v8f;

#define HID    128
#define NFEAT  98
#define KC1    64
#define KC2    64
#define NBLOCK 3
#define CUTOFF_R 8.0f
#define MAXG   50

// Fast sigmoid: single v_exp_f32 + v_rcp_f32 (avoid IEEE div expansion).
__device__ __forceinline__ float sigmoidf_(float v){
  return __builtin_amdgcn_rcpf(1.0f + __expf(-v));
}
__device__ __forceinline__ float eluf_(float v){ return v > 0.0f ? v : (__expf(v) - 1.0f); }

__device__ __forceinline__ v8f wmma_f16(v16h a, v16h b, v8f c){
  // D = A(16x32 f16) * B(32x16 f16) + C(16x16 f32)
  return __builtin_amdgcn_wmma_f32_16x16x32_f16(false, a, false, b, (short)0, c, false, false);
}

// A fragment from an LDS tile stored row-major [16][ldk] halves.
// Lane layout (ISA 7.12.2, 16-bit A 16x32): M = lane&15,
// halves 0..7 -> K = kt*32 + ((lane&16)>>1) + i,  halves 8..15 -> +16.
__device__ __forceinline__ v16h load_A_lds(const _Float16* tile, int lane, int kt, int ldk){
  int m  = lane & 15;
  int k0 = kt*32 + ((lane & 16) >> 1);
  const _Float16* p = tile + m*ldk + k0;
  v8h lo = *(const v8h*)(p);
  v8h hi = *(const v8h*)(p + 16);
  v16h a;
  #pragma unroll
  for (int i = 0; i < 8; ++i){ a[i] = lo[i]; a[i+8] = hi[i]; }
  return a;
}

// B fragment from row-major f16 weights W[n][k] (out = sum_k A[m][k]*W[n][k]).
// Lane layout: N = nbase + (lane&15), halves 0..7 -> K = kt*32 + ((lane&16)>>1) + i,
// halves 8..15 -> +16 (same K order as load_A_lds).
__device__ __forceinline__ v16h load_B_row(const _Float16* __restrict__ W,
                                           int lane, int nbase, int kt, int ldk){
  int n  = nbase + (lane & 15);
  int k0 = kt*32 + ((lane & 16) >> 1);
  const _Float16* p = W + n*ldk + k0;
  v8h lo = *(const v8h*)(p);
  v8h hi = *(const v8h*)(p + 16);
  v16h b;
  #pragma unroll
  for (int i = 0; i < 8; ++i){ b[i] = lo[i]; b[i+8] = hi[i]; }
  return b;
}

// C/D mapping: lane = N + 16*(M>>3), vgpr v = M&7  =>  M = ((lane>>4)<<3)+v, N = lane&15.

// ------------------------------- utility kernels ---------------------------

__global__ void k_cvt_f16(const float* __restrict__ s, _Float16* __restrict__ d, int n){
  int i = blockIdx.x*blockDim.x + threadIdx.x;
  if (i < n) d[i] = (_Float16)s[i];
}

__global__ void k_pad_wemb(const float* __restrict__ s, _Float16* __restrict__ d){
  int i = blockIdx.x*blockDim.x + threadIdx.x;   // 128*128
  if (i >= HID*HID) return;
  int r = i >> 7, c = i & 127;
  d[i] = (c < NFEAT) ? (_Float16)s[r*NFEAT + c] : (_Float16)0.0f;
}

__global__ void k_zero_f32(float* __restrict__ p, int n){
  int i = blockIdx.x*blockDim.x + threadIdx.x;
  if (i < n) p[i] = 0.0f;
}

// ------------------------------- embedding ---------------------------------
// x = sigmoid(nodes @ W_emb.T), K padded 98->128. Writes both ping-pong bufs.
__global__ __launch_bounds__(256) void k_embed(const float* __restrict__ nodes,
                                               const _Float16* __restrict__ WembH,
                                               float* __restrict__ x0,
                                               float* __restrict__ x1, int nNodes){
  __shared__ __align__(32) _Float16 aH[16*HID];
  int tid = threadIdx.x;
  int r = tid >> 4, q = tid & 15;
  int node = blockIdx.x*16 + r;
  bool nv = node < nNodes;
  #pragma unroll
  for (int j = 0; j < 8; ++j){
    int c = q*8 + j;
    float v = (nv && c < NFEAT) ? nodes[node*NFEAT + c] : 0.0f;
    aH[r*HID + c] = (_Float16)v;
  }
  __syncthreads();

  int wave = tid >> 5, lane = tid & 31, nbase = wave*16;
  v8f acc = {};
  #pragma unroll
  for (int kt = 0; kt < 4; ++kt){
    v16h A = load_A_lds(aH, lane, kt, HID);
    v16h B = load_B_row(WembH, lane, nbase, kt, HID);
    acc = wmma_f16(A, B, acc);
  }
  int Mb = ((lane >> 4) << 3);
  int N  = nbase + (lane & 15);
  int ndb = blockIdx.x*16 + Mb;
  if (ndb + 8 <= nNodes){                       // full-tile fast path
    #pragma unroll
    for (int v = 0; v < 8; ++v){
      float s = sigmoidf_(acc[v]);
      size_t idx = (size_t)(ndb + v)*HID + N;
      x0[idx] = s; x1[idx] = s;
    }
  } else {
    #pragma unroll
    for (int v = 0; v < 8; ++v){
      if (ndb + v < nNodes){
        float s = sigmoidf_(acc[v]);
        size_t idx = (size_t)(ndb + v)*HID + N;
        x0[idx] = s; x1[idx] = s;
      }
    }
  }
}

// ------------------------------- edge block --------------------------------
// fe = [ni | nj | (ni-nj)/rij] ; z = sig(fe@Wg.T)*elu(fe@Wm.T)*(z1+z2)*mask
// scatter-add into x_out[src] (x_out is a copy of x_in -> matches .at[].add).
__global__ __launch_bounds__(256) void k_edge(
    const float* __restrict__ x_in, float* __restrict__ x_out,
    const int* __restrict__ esrc, const int* __restrict__ etgt,
    const float* __restrict__ edist,
    const float* __restrict__ cset, const float* __restrict__ pwave,
    const _Float16* __restrict__ WgH,  const _Float16* __restrict__ WmH,
    const _Float16* __restrict__ W1vH, const _Float16* __restrict__ W2vH,
    const _Float16* __restrict__ W2vgH, int nEdges){
  __shared__ __align__(32) _Float16 feH[16*(3*HID)];   // 12 KB
  __shared__ __align__(32) _Float16 csH[16*KC1];
  __shared__ __align__(32) _Float16 pwH[16*KC2];
  __shared__ __align__(32) _Float16 pwgH[16*KC2];
  __shared__ float maskS[16];
  __shared__ int   srcS[16];

  int tid = threadIdx.x;
  int r = tid >> 4, q = tid & 15;
  int e = blockIdx.x*16 + r;
  bool ev = e < nEdges;
  int si = ev ? esrc[e] : 0;
  int ti = ev ? etgt[e] : 0;
  float rij = ev ? edist[e] : 1.0f;
  float invr = __builtin_amdgcn_rcpf(rij);
  if (q == 0){
    srcS[r]  = si;
    maskS[r] = (ev && rij < CUTOFF_R) ? 1.0f : 0.0f;
  }
  const float* xi = x_in + (size_t)si*HID;
  const float* xj = x_in + (size_t)ti*HID;
  #pragma unroll
  for (int j = 0; j < 2; ++j){
    int c = q*8 + j*4;
    float4 a = ev ? *(const float4*)(xi + c) : make_float4(0.f,0.f,0.f,0.f);
    float4 b = ev ? *(const float4*)(xj + c) : make_float4(0.f,0.f,0.f,0.f);
    _Float16* f0 = feH + r*(3*HID) + c;
    f0[0]=(_Float16)a.x; f0[1]=(_Float16)a.y; f0[2]=(_Float16)a.z; f0[3]=(_Float16)a.w;
    _Float16* f1 = f0 + HID;
    f1[0]=(_Float16)b.x; f1[1]=(_Float16)b.y; f1[2]=(_Float16)b.z; f1[3]=(_Float16)b.w;
    _Float16* f2 = f0 + 2*HID;
    f2[0]=(_Float16)((a.x-b.x)*invr); f2[1]=(_Float16)((a.y-b.y)*invr);
    f2[2]=(_Float16)((a.z-b.z)*invr); f2[3]=(_Float16)((a.w-b.w)*invr);
  }
  #pragma unroll
  for (int j = 0; j < 4; ++j){
    int c = q*4 + j;
    csH[r*KC1 + c] = (_Float16)(ev ? cset[(size_t)e*KC1 + c] : 0.0f);
    pwH[r*KC2 + c] = (_Float16)(ev ? pwave[(size_t)e*KC2 + c] : 0.0f);
  }
  __syncthreads();

  int wave = tid >> 5, lane = tid & 31, nbase = wave*16;

  // big GEMMs: fe @ Wg.T and fe @ Wm.T   (K = 384 -> 12 steps)
  v8f accG = {}, accM = {};
  #pragma unroll
  for (int kt = 0; kt < 12; ++kt){
    v16h A  = load_A_lds(feH, lane, kt, 3*HID);
    v16h Bg = load_B_row(WgH, lane, nbase, kt, 3*HID);
    accG = wmma_f16(A, Bg, accG);
    v16h Bm = load_B_row(WmH, lane, nbase, kt, 3*HID);
    accM = wmma_f16(A, Bm, accM);
  }

  // z1 = combine_sets @ W1v.T
  v8f accZ = {};
  #pragma unroll
  for (int kt = 0; kt < 2; ++kt){
    v16h A = load_A_lds(csH, lane, kt, KC1);
    v16h B = load_B_row(W1vH, lane, nbase, kt, KC1);
    accZ = wmma_f16(A, B, accZ);
  }

  // pw_gate = sigmoid(pw @ W2vg.T); waves 0..3 own 16 gate cols each
  if (wave < 4){
    int gb = wave*16;
    v8f accP = {};
    #pragma unroll
    for (int kt = 0; kt < 2; ++kt){
      v16h A = load_A_lds(pwH, lane, kt, KC2);
      v16h B = load_B_row(W2vgH, lane, gb, kt, KC2);
      accP = wmma_f16(A, B, accP);
    }
    #pragma unroll
    for (int v = 0; v < 8; ++v){
      int M = ((lane >> 4) << 3) + v;
      int N = gb + (lane & 15);
      float g  = sigmoidf_(accP[v]);
      float pv = (float)pwH[M*KC2 + N];
      pwgH[M*KC2 + N] = (_Float16)(pv * g);
    }
  }
  __syncthreads();

  // z2 = (pw * gate) @ W2v.T
  #pragma unroll
  for (int kt = 0; kt < 2; ++kt){
    v16h A = load_A_lds(pwgH, lane, kt, KC2);
    v16h B = load_B_row(W2vH, lane, nbase, kt, KC2);
    accZ = wmma_f16(A, B, accZ);
  }

  // epilogue + scatter-add; skip atomics for masked-out / OOB edges (z == 0)
  int Mb = ((lane >> 4) << 3);
  int N  = nbase + (lane & 15);
  #pragma unroll
  for (int v = 0; v < 8; ++v){
    int M = Mb + v;
    float mk = maskS[M];
    if (mk != 0.0f){
      float z = sigmoidf_(accG[v]) * eluf_(accM[v]) * accZ[v];
      atomicAdd(&x_out[(size_t)srcS[M]*HID + N], z);
    }
  }
}

// ------------------------------- node block --------------------------------
// zp = elu(x@Wp1.T) * (x@Wp2.T); pooled[g] += zp; x_next = elu(x@Wpsi.T)
// Writes x_next into xn0 and (if non-null) xn1 = pre-copied buffer for next
// block's edge scatter. In-place write of x_cur rows is race-free: each row
// is touched only by its owning workgroup.
__global__ __launch_bounds__(256) void k_node(
    const float* __restrict__ x_cur, float* __restrict__ xn0,
    float* __restrict__ xn1,
    const _Float16* __restrict__ Wp1H, const _Float16* __restrict__ Wp2H,
    const _Float16* __restrict__ WpsH,
    const int* __restrict__ gidx, float* __restrict__ pooled, int nNodes){
  __shared__ __align__(32) _Float16 xH[16*HID];
  __shared__ int gS[16];
  int tid = threadIdx.x;
  int r = tid >> 4, q = tid & 15;
  int node = blockIdx.x*16 + r;
  bool nv = node < nNodes;
  const float* xr = x_cur + (size_t)node*HID;
  #pragma unroll
  for (int j = 0; j < 2; ++j){
    int c = q*8 + j*4;
    float4 a = nv ? *(const float4*)(xr + c) : make_float4(0.f,0.f,0.f,0.f);
    _Float16* p = xH + r*HID + c;
    p[0]=(_Float16)a.x; p[1]=(_Float16)a.y; p[2]=(_Float16)a.z; p[3]=(_Float16)a.w;
  }
  if (q == 0) gS[r] = nv ? gidx[node] : 0;
  __syncthreads();

  int wave = tid >> 5, lane = tid & 31, nbase = wave*16;
  v8f a1 = {}, a2 = {}, as = {};
  #pragma unroll
  for (int kt = 0; kt < 4; ++kt){
    v16h A  = load_A_lds(xH, lane, kt, HID);
    v16h B1 = load_B_row(Wp1H, lane, nbase, kt, HID);
    a1 = wmma_f16(A, B1, a1);
    v16h B2 = load_B_row(Wp2H, lane, nbase, kt, HID);
    a2 = wmma_f16(A, B2, a2);
    v16h B3 = load_B_row(WpsH, lane, nbase, kt, HID);
    as = wmma_f16(A, B3, as);
  }
  int Mb = ((lane >> 4) << 3);
  int N  = nbase + (lane & 15);
  int ndb = blockIdx.x*16 + Mb;
  if (ndb + 8 <= nNodes){                       // full-tile fast path
    #pragma unroll
    for (int v = 0; v < 8; ++v){
      float zp = eluf_(a1[v]) * a2[v];
      atomicAdd(&pooled[(size_t)gS[Mb + v]*HID + N], zp);
      float xn = eluf_(as[v]);
      size_t idx = (size_t)(ndb + v)*HID + N;
      xn0[idx] = xn;
      if (xn1) xn1[idx] = xn;
    }
  } else {
    #pragma unroll
    for (int v = 0; v < 8; ++v){
      if (ndb + v < nNodes){
        float zp = eluf_(a1[v]) * a2[v];
        atomicAdd(&pooled[(size_t)gS[Mb + v]*HID + N], zp);
        float xn = eluf_(as[v]);
        size_t idx = (size_t)(ndb + v)*HID + N;
        xn0[idx] = xn;
        if (xn1) xn1[idx] = xn;
      }
    }
  }
}

// ------------------------------- readout MLP -------------------------------
__global__ __launch_bounds__(256) void k_readout(
    const float* __restrict__ pooled,
    const float* __restrict__ Wlr1, const float* __restrict__ Wlr2,
    const float* __restrict__ Wlr3, float* __restrict__ y, int nG){
  __shared__ float h1[MAXG*64];
  __shared__ float h2[MAXG*32];
  int tid = threadIdx.x;
  if (nG > MAXG) nG = MAXG;
  for (int idx = tid; idx < nG*64; idx += blockDim.x){
    int g = idx >> 6, j = idx & 63;
    float s = 0.0f;
    for (int k = 0; k < HID; ++k) s += pooled[g*HID + k] * Wlr1[j*HID + k];
    h1[idx] = eluf_(s);
  }
  __syncthreads();
  for (int idx = tid; idx < nG*32; idx += blockDim.x){
    int g = idx >> 5, j = idx & 31;
    float s = 0.0f;
    for (int k = 0; k < 64; ++k) s += h1[g*64 + k] * Wlr2[j*64 + k];
    h2[idx] = eluf_(s);
  }
  __syncthreads();
  for (int g = tid; g < nG; g += blockDim.x){
    float s = 0.0f;
    for (int k = 0; k < 32; ++k) s += h2[g*32 + k] * Wlr3[k];
    y[g] = s;
  }
}

// ------------------------------- host orchestration ------------------------

extern "C" void kernel_launch(void* const* d_in, const int* in_sizes, int n_in,
                              void* d_out, int out_size, void* d_ws, size_t ws_size,
                              hipStream_t stream){
  (void)n_in; (void)ws_size; (void)out_size;
  const float* nodes = (const float*)d_in[0];
  const int*   esrc  = (const int*)  d_in[1];
  const int*   etgt  = (const int*)  d_in[2];
  const float* edist = (const float*)d_in[3];
  const int*   gidx  = (const int*)  d_in[4];
  const float* cset  = (const float*)d_in[6];
  const float* pwave = (const float*)d_in[7];
  const float* Wemb  = (const float*)d_in[8];
  const float* W1v   = (const float*)d_in[9];
  const float* W2v   = (const float*)d_in[10];
  const float* W2vg  = (const float*)d_in[11];
  const float* Wg    = (const float*)d_in[12];
  const float* Wm    = (const float*)d_in[13];
  const float* Wpsi  = (const float*)d_in[14];
  const float* Wp1   = (const float*)d_in[15];
  const float* Wp2   = (const float*)d_in[16];
  const float* Wlr1  = (const float*)d_in[17];
  const float* Wlr2  = (const float*)d_in[18];
  const float* Wlr3  = (const float*)d_in[19];
  float* y = (float*)d_out;

  const int nEdges  = in_sizes[1];
  const int nNodes  = in_sizes[4];
  const int nGraphs = in_sizes[5];

  // carve workspace (256-byte aligned slices)
  char* p = (char*)d_ws;
  auto carve = [&](size_t bytes)->char*{
    char* r = p; p += (bytes + 255) & ~(size_t)255; return r;
  };
  float*    x_a    = (float*)   carve((size_t)nNodes*HID*4);
  float*    x_b    = (float*)   carve((size_t)nNodes*HID*4);
  float*    pooled = (float*)   carve((size_t)nGraphs*HID*4);
  _Float16* WembH  = (_Float16*)carve((size_t)HID*HID*2);
  _Float16* WgH    = (_Float16*)carve((size_t)NBLOCK*HID*3*HID*2);
  _Float16* WmH    = (_Float16*)carve((size_t)NBLOCK*HID*3*HID*2);
  _Float16* W1vH   = (_Float16*)carve((size_t)NBLOCK*HID*KC1*2);
  _Float16* W2vH   = (_Float16*)carve((size_t)NBLOCK*HID*KC2*2);
  _Float16* W2vgH  = (_Float16*)carve((size_t)NBLOCK*KC2*KC2*2);
  _Float16* WpsH   = (_Float16*)carve((size_t)NBLOCK*HID*HID*2);
  _Float16* Wp1H   = (_Float16*)carve((size_t)NBLOCK*HID*HID*2);
  _Float16* Wp2H   = (_Float16*)carve((size_t)NBLOCK*HID*HID*2);

  auto cvt = [&](const float* s, _Float16* d, int n){
    k_cvt_f16<<<(n + 255)/256, 256, 0, stream>>>(s, d, n);
  };
  cvt(Wg,   WgH,   NBLOCK*HID*3*HID);
  cvt(Wm,   WmH,   NBLOCK*HID*3*HID);
  cvt(W1v,  W1vH,  NBLOCK*HID*KC1);
  cvt(W2v,  W2vH,  NBLOCK*HID*KC2);
  cvt(W2vg, W2vgH, NBLOCK*KC2*KC2);
  cvt(Wpsi, WpsH,  NBLOCK*HID*HID);
  cvt(Wp1,  Wp1H,  NBLOCK*HID*HID);
  cvt(Wp2,  Wp2H,  NBLOCK*HID*HID);
  k_pad_wemb<<<(HID*HID + 255)/256, 256, 0, stream>>>(Wemb, WembH);
  k_zero_f32<<<(nGraphs*HID + 255)/256, 256, 0, stream>>>(pooled, nGraphs*HID);

  const int nodeTiles = (nNodes + 15)/16;
  const int edgeTiles = (nEdges + 15)/16;

  // embed writes x_a and its copy x_b (edge kernel gathers x_a, scatters x_b)
  k_embed<<<nodeTiles, 256, 0, stream>>>(nodes, WembH, x_a, x_b, nNodes);

  for (int b = 0; b < NBLOCK; ++b){
    k_edge<<<edgeTiles, 256, 0, stream>>>(
        x_a, x_b, esrc, etgt, edist, cset, pwave,
        WgH  + (size_t)b*HID*3*HID, WmH + (size_t)b*HID*3*HID,
        W1vH + (size_t)b*HID*KC1,   W2vH + (size_t)b*HID*KC2,
        W2vgH+ (size_t)b*KC2*KC2,   nEdges);
    // node reads x_b (post-scatter), writes elu(psi) into x_a and, except on
    // the last block, also into x_b (pre-copy for next block's scatter).
    k_node<<<nodeTiles, 256, 0, stream>>>(
        x_b, x_a, (b + 1 < NBLOCK) ? x_b : (float*)nullptr,
        Wp1H + (size_t)b*HID*HID, Wp2H + (size_t)b*HID*HID,
        WpsH + (size_t)b*HID*HID, gidx, pooled, nNodes);
  }

  k_readout<<<1, 256, 0, stream>>>(pooled, Wlr1, Wlr2, Wlr3, y, nGraphs);
}
